// Decoder_87737591922939
// MI455X (gfx1250) — matrix operand order repaired and checked by reading, not compiled, not run on hardware
//
#include <hip/hip_runtime.h>
#include <hip/hip_bf16.h>
#include <math.h>

// ---------------- problem constants ----------------
#define EMB   512
#define ENC   1024
#define HID   1024
#define VSZ   32000
#define INTER 2160
#define KPAD  2176      // INTER padded to multiple of 32 (K of logits GEMM)
#define NPAD  2176      // INTER padded to multiple of 64 (N of inter GEMM)
#define BATCH 32
#define SENC  64
#define TSTEP 63        // T_DEC - 1 scan steps

typedef __bf16 bf16_t;
typedef bf16_t bf16x8  __attribute__((ext_vector_type(8)));
typedef bf16_t bf16x16 __attribute__((ext_vector_type(16)));
typedef float  f32x8   __attribute__((ext_vector_type(8)));

__device__ __forceinline__ unsigned short f2bf(float f) {
  union { float f; unsigned int u; } v; v.f = f;
  unsigned int r = v.u + 0x7FFFu + ((v.u >> 16) & 1u);   // round-to-nearest-even
  return (unsigned short)(r >> 16);
}

// =====================================================================
// WMMA GEMM:  C[M,N] = A[M,K](bf16) * W[N,K](bf16 row-major)^T
// Each wave owns a 32(M) x 64(N) strip: 2 A-tiles x 4 B-tiles = 8
// accumulators; every B tile is reused for both A tiles (halves B traffic).
// Requires M % 32 == 0 and N % 64 == 0. All accumulator indices are
// compile-time constants (no v_movrel).
// A (16x32): lane row = m + (lane&15); elems 0..7 @ k0+(lane>>4)*16,
//            elems 8..15 @ k0+8+(lane>>4)*16   (two 16B loads)
// B (32x16): lane col = n + (lane&15); 16 contiguous elems @ k0+(lane>>4)*16
//            of weight row n                    (one 32B load)
// D: c[r] -> M = mtile + r + 8*(lane>>4), N = ntile + (lane&15)
// act: 0 = +bias, 1 = tanh(x+bias). Cf (f32) and/or Cb (bf16) outputs.
// =====================================================================
__global__ void __launch_bounds__(128)
wmma_gemm(const unsigned short* __restrict__ A, int lda,
          const unsigned short* __restrict__ W, int ldw,
          const float* __restrict__ bias,
          float* __restrict__ Cf, unsigned short* __restrict__ Cb, long long ldc,
          int M, int N, int K, int act)
{
  const int wave = threadIdx.x >> 5;
  const int lane = threadIdx.x & 31;
  const int n0 = (blockIdx.x * 4 + wave) * 64;
  const int m0 = blockIdx.y * 32;
  if (n0 >= N) return;                    // wave-uniform: EXEC stays all-1s
  const int lr = lane & 15;
  const int lh = lane >> 4;               // 0 or 1

  const unsigned short* arow0 = A + (size_t)(m0 + lr) * lda + lh * 16;
  const unsigned short* arow1 = arow0 + (size_t)16 * lda;
  const unsigned short* w0 = W + (size_t)(n0      + lr) * ldw + lh * 16;
  const unsigned short* w1 = W + (size_t)(n0 + 16 + lr) * ldw + lh * 16;
  const unsigned short* w2 = W + (size_t)(n0 + 32 + lr) * ldw + lh * 16;
  const unsigned short* w3 = W + (size_t)(n0 + 48 + lr) * ldw + lh * 16;

  f32x8 c00 = {}, c01 = {}, c02 = {}, c03 = {};
  f32x8 c10 = {}, c11 = {}, c12 = {}, c13 = {};

  for (int k0 = 0; k0 < K; k0 += 32) {
    bf16x8 a0lo = *reinterpret_cast<const bf16x8*>(arow0 + k0);
    bf16x8 a0hi = *reinterpret_cast<const bf16x8*>(arow0 + k0 + 8);
    bf16x8 a1lo = *reinterpret_cast<const bf16x8*>(arow1 + k0);
    bf16x8 a1hi = *reinterpret_cast<const bf16x8*>(arow1 + k0 + 8);
    bf16x16 a0 = __builtin_shufflevector(a0lo, a0hi,
                   0,1,2,3,4,5,6,7,8,9,10,11,12,13,14,15);
    bf16x16 a1 = __builtin_shufflevector(a1lo, a1hi,
                   0,1,2,3,4,5,6,7,8,9,10,11,12,13,14,15);
    bf16x16 b0 = *reinterpret_cast<const bf16x16*>(w0 + k0);
    bf16x16 b1 = *reinterpret_cast<const bf16x16*>(w1 + k0);
    bf16x16 b2 = *reinterpret_cast<const bf16x16*>(w2 + k0);
    bf16x16 b3 = *reinterpret_cast<const bf16x16*>(w3 + k0);
    c00 = __builtin_amdgcn_wmma_f32_16x16x32_bf16(false, a0, false, b0, (short)0, c00, false, false);
    c10 = __builtin_amdgcn_wmma_f32_16x16x32_bf16(false, a1, false, b0, (short)0, c10, false, false);
    c01 = __builtin_amdgcn_wmma_f32_16x16x32_bf16(false, a0, false, b1, (short)0, c01, false, false);
    c11 = __builtin_amdgcn_wmma_f32_16x16x32_bf16(false, a1, false, b1, (short)0, c11, false, false);
    c02 = __builtin_amdgcn_wmma_f32_16x16x32_bf16(false, a0, false, b2, (short)0, c02, false, false);
    c12 = __builtin_amdgcn_wmma_f32_16x16x32_bf16(false, a1, false, b2, (short)0, c12, false, false);
    c03 = __builtin_amdgcn_wmma_f32_16x16x32_bf16(false, a0, false, b3, (short)0, c03, false, false);
    c13 = __builtin_amdgcn_wmma_f32_16x16x32_bf16(false, a1, false, b3, (short)0, c13, false, false);
  }

  f32x8 accs[2][4] = { { c00, c01, c02, c03 }, { c10, c11, c12, c13 } };
  #pragma unroll
  for (int j = 0; j < 4; ++j) {
    int n = n0 + 16 * j + lr;
    float bv = bias ? bias[n] : 0.0f;
    #pragma unroll
    for (int h = 0; h < 2; ++h) {
      int mbase = m0 + 16 * h + lh * 8;
      #pragma unroll
      for (int r = 0; r < 8; ++r) {
        float x = accs[h][j][r] + bv;
        if (act == 1) x = tanhf(x);
        long long idx = (long long)(mbase + r) * ldc + n;
        if (Cf) Cf[idx] = x;
        if (Cb) Cb[idx] = f2bf(x);
      }
    }
  }
}

// ---------------- staging kernels (run once per launch) ----------------
__global__ void cvt_bf16(const float* __restrict__ src, unsigned short* __restrict__ dst,
                         long long rows, int scols, int dstride, int doff) {
  long long total = rows * (long long)scols;
  for (long long i = blockIdx.x * (long long)blockDim.x + threadIdx.x;
       i < total; i += (long long)gridDim.x * blockDim.x) {
    long long r = i / scols; int c = (int)(i - r * scols);
    dst[r * dstride + doff + c] = f2bf(src[i]);
  }
}

__global__ void zero_bf(unsigned short* __restrict__ dst, long long rows,
                        int stride, int col0, int ncols) {
  long long total = rows * (long long)ncols;
  for (long long i = blockIdx.x * (long long)blockDim.x + threadIdx.x;
       i < total; i += (long long)gridDim.x * blockDim.x) {
    long long r = i / ncols; int c = (int)(i - r * ncols);
    dst[r * stride + col0 + c] = 0;
  }
}

// dst[n*K + k] = bf16(src[k*N + n])   (stage W^T for x@W GEMMs)
__global__ void transpose_cvt(const float* __restrict__ src, unsigned short* __restrict__ dst,
                              int K, int N) {
  long long total = (long long)K * N;
  for (long long i = blockIdx.x * (long long)blockDim.x + threadIdx.x;
       i < total; i += (long long)gridDim.x * blockDim.x) {
    int n = (int)(i / K); int k = (int)(i - (long long)n * K);
    dst[i] = f2bf(src[(long long)k * N + n]);
  }
}

__global__ void vec_add(const float* a, const float* b, float* o, int n) {
  int i = blockIdx.x * blockDim.x + threadIdx.x;
  if (i < n) o[i] = a[i] + b[i];
}

// copy n floats then zero-pad to ntot
__global__ void copy_pad_f32(const float* __restrict__ src, float* __restrict__ dst,
                             int n, int ntot) {
  int i = blockIdx.x * blockDim.x + threadIdx.x;
  if (i < ntot) dst[i] = (i < n) ? src[i] : 0.0f;
}

// embs_bf[(t*32+b)*512 + e] = bf16(emb_table[y[b*64+t]*512 + e]),  t in [0,63)
__global__ void embed_gather(const float* __restrict__ table, const int* __restrict__ y,
                             unsigned short* __restrict__ embs) {
  long long total = (long long)TSTEP * BATCH * EMB;
  for (long long i = blockIdx.x * (long long)blockDim.x + threadIdx.x;
       i < total; i += (long long)gridDim.x * blockDim.x) {
    int e = (int)(i & (EMB - 1));
    long long tb = i >> 9;
    int b = (int)(tb & (BATCH - 1));
    int t = (int)(tb >> 5);
    int tok = y[b * 64 + t];
    embs[i] = f2bf(table[(long long)tok * EMB + e]);
  }
}

__global__ void init_state(const float* __restrict__ dh, const float* __restrict__ dc,
                           float* h0, float* c0, float* h1, float* c1,
                           unsigned short* h1b) {
  int i = blockIdx.x * blockDim.x + threadIdx.x;   // 32768
  float h1v = dh[BATCH * HID + i];
  h0[i] = dh[i];  h1[i] = h1v;
  c0[i] = dc[i];  c1[i] = dc[BATCH * HID + i];
  h1b[i] = f2bf(h1v);
}

// ---------------- per-step kernels ----------------
// A0 = [emb(512) | ctx(1024 by softmax) | h0_prev(1024)]   stride 2560
// A1 = [h0_new(by lstm0) | h1_prev(1024)]                   stride 2048
// lin_in = [h1_new(by lstm1) | ctx(by softmax) | emb(512)]  stride 2560
__global__ void prep_step(const unsigned short* __restrict__ emb_t,
                          const float* __restrict__ h0, const float* __restrict__ h1,
                          unsigned short* A0, unsigned short* lin_in, unsigned short* A1) {
  int i = blockIdx.x * blockDim.x + threadIdx.x;   // 32768
  int b = i >> 10, j = i & 1023;
  if (j < EMB) {
    unsigned short e = emb_t[b * EMB + j];
    A0[(size_t)b * 2560 + j] = e;
    lin_in[(size_t)b * 2560 + 2048 + j] = e;
  }
  A0[(size_t)b * 2560 + 1536 + j] = f2bf(h0[i]);
  A1[(size_t)b * 2048 + 1024 + j] = f2bf(h1[i]);
}

// score[b][s] = sum_h tanh(enc_proj[b][s][h] + q[b][h]) * v[h]
__global__ void __launch_bounds__(128)
attn_score(const float* __restrict__ enc_proj, const float* __restrict__ q,
           const float* __restrict__ v, float* __restrict__ score) {
  int s = blockIdx.x, b = blockIdx.y;
  const float* ep = enc_proj + ((size_t)b * SENC + s) * HID;
  const float* qb = q + (size_t)b * HID;
  float acc = 0.0f;
  for (int h = threadIdx.x; h < HID; h += 128)
    acc += tanhf(ep[h] + qb[h]) * v[h];
  __shared__ float red[128];
  red[threadIdx.x] = acc; __syncthreads();
  for (int off = 64; off; off >>= 1) {
    if (threadIdx.x < off) red[threadIdx.x] += red[threadIdx.x + off];
    __syncthreads();
  }
  if (threadIdx.x == 0) score[b * SENC + s] = red[0];
}

// masked softmax over s, write attw, ctx -> bf16 into A0[:,512:1536] & lin_in[:,1024:2048]
__global__ void __launch_bounds__(256)
attn_softmax_ctx(const float* __restrict__ score, const int* __restrict__ mask,
                 const float* __restrict__ enc_out, float* __restrict__ attw,
                 unsigned short* __restrict__ A0, unsigned short* __restrict__ lin_in) {
  int b = blockIdx.x, tid = threadIdx.x;
  __shared__ float w[SENC];
  if (tid < SENC)
    w[tid] = mask[b * SENC + tid] ? score[b * SENC + tid] : -1.0e9f;
  __syncthreads();
  if (tid == 0) {
    float m = w[0];
    for (int s = 1; s < SENC; ++s) m = fmaxf(m, w[s]);
    float sum = 0.0f;
    for (int s = 0; s < SENC; ++s) { float e = __expf(w[s] - m); w[s] = e; sum += e; }
    float inv = 1.0f / sum;
    for (int s = 0; s < SENC; ++s) w[s] *= inv;
  }
  __syncthreads();
  if (tid < SENC) attw[b * SENC + tid] = w[tid];
  for (int d = tid; d < ENC; d += 256) {
    float acc = 0.0f;
    const float* eo = enc_out + ((size_t)b * SENC) * ENC + d;
    for (int s = 0; s < SENC; ++s) acc += w[s] * eo[(size_t)s * ENC];
    unsigned short cb = f2bf(acc);
    A0[(size_t)b * 2560 + 512 + d]      = cb;
    lin_in[(size_t)b * 2560 + 1024 + d] = cb;
  }
}

// LSTM pointwise: gates[b][4096] (i|f|g|o) + bsum -> new c,h (f32) + bf16 copies
__global__ void __launch_bounds__(256)
lstm_update(const float* __restrict__ gates, const float* __restrict__ bsum,
            float* __restrict__ c, float* __restrict__ h,
            unsigned short* bd1, int bd1_stride,
            unsigned short* bd2, int bd2_stride) {
  int i = blockIdx.x * blockDim.x + threadIdx.x;   // 32768
  int b = i >> 10, j = i & 1023;
  const float* g = gates + (size_t)b * 4096;
  float gi = g[j]        + bsum[j];
  float gf = g[1024 + j] + bsum[1024 + j];
  float gg = g[2048 + j] + bsum[2048 + j];
  float go = g[3072 + j] + bsum[3072 + j];
  float si = 1.0f / (1.0f + __expf(-gi));
  float sf = 1.0f / (1.0f + __expf(-gf));
  float so = 1.0f / (1.0f + __expf(-go));
  float cn = sf * c[i] + si * tanhf(gg);
  float hn = so * tanhf(cn);
  c[i] = cn; h[i] = hn;
  unsigned short hb = f2bf(hn);
  if (bd1) bd1[(size_t)b * bd1_stride + j] = hb;
  if (bd2) bd2[(size_t)b * bd2_stride + j] = hb;
}

// =====================================================================
extern "C" void kernel_launch(void* const* d_in, const int* in_sizes, int n_in,
                              void* d_out, int out_size, void* d_ws, size_t ws_size,
                              hipStream_t stream) {
  (void)in_sizes; (void)n_in; (void)out_size; (void)ws_size;
  const float* enc_output = (const float*)d_in[0];
  const int*   enc_mask   = (const int*)  d_in[1];
  const int*   y_input    = (const int*)  d_in[2];
  const float* dec_h0     = (const float*)d_in[3];
  const float* dec_c0     = (const float*)d_in[4];
  const float* emb_table  = (const float*)d_in[5];
  const float* We         = (const float*)d_in[6];
  const float* Wd         = (const float*)d_in[7];
  const float* vvec       = (const float*)d_in[8];
  const float* W_ih0      = (const float*)d_in[9];
  const float* W_hh0      = (const float*)d_in[10];
  const float* b_ih0      = (const float*)d_in[11];
  const float* b_hh0      = (const float*)d_in[12];
  const float* W_ih1      = (const float*)d_in[13];
  const float* W_hh1      = (const float*)d_in[14];
  const float* b_ih1      = (const float*)d_in[15];
  const float* b_hh1      = (const float*)d_in[16];
  const float* out_W      = (const float*)d_in[17];
  const float* out_b      = (const float*)d_in[18];
  const float* sm_W       = (const float*)d_in[19];
  const float* sm_b       = (const float*)d_in[20];

  float* outF = (float*)d_out;
  const long long OUT0 = (long long)BATCH * TSTEP * VSZ;   // logits, then attw

  // -------- workspace carving (256B aligned) --------
  size_t cur = 0;
  auto carve = [&](size_t bytes) -> char* {
    cur = (cur + 255) & ~(size_t)255;
    char* p = (char*)d_ws + cur; cur += bytes; return p;
  };
  unsigned short* smW_bf   = (unsigned short*)carve((size_t)VSZ  * KPAD * 2);
  unsigned short* outW_bf  = (unsigned short*)carve((size_t)NPAD * 2560 * 2);
  unsigned short* Wcat0_bf = (unsigned short*)carve((size_t)4096 * 2560 * 2);
  unsigned short* Wcat1_bf = (unsigned short*)carve((size_t)4096 * 2048 * 2);
  unsigned short* WeT_bf   = (unsigned short*)carve((size_t)HID  * ENC  * 2);
  unsigned short* WdT_bf   = (unsigned short*)carve((size_t)HID  * HID  * 2);
  unsigned short* encO_bf  = (unsigned short*)carve((size_t)BATCH* SENC * ENC * 2);
  unsigned short* embs_bf  = (unsigned short*)carve((size_t)TSTEP* BATCH* EMB * 2);
  unsigned short* A0_bf    = (unsigned short*)carve((size_t)BATCH* 2560 * 2);
  unsigned short* A1_bf    = (unsigned short*)carve((size_t)BATCH* 2048 * 2);
  unsigned short* lin_bf   = (unsigned short*)carve((size_t)BATCH* 2560 * 2);
  unsigned short* inter_bf = (unsigned short*)carve((size_t)BATCH* KPAD * 2);
  unsigned short* h1_bf    = (unsigned short*)carve((size_t)BATCH* HID  * 2);
  float* enc_proj = (float*)carve((size_t)BATCH * SENC * HID * 4);
  float* qbuf     = (float*)carve((size_t)BATCH * HID * 4);
  float* score    = (float*)carve((size_t)BATCH * SENC * 4);
  float* gates    = (float*)carve((size_t)BATCH * 4096 * 4);
  float* h0f = (float*)carve((size_t)BATCH * HID * 4);
  float* c0f = (float*)carve((size_t)BATCH * HID * 4);
  float* h1f = (float*)carve((size_t)BATCH * HID * 4);
  float* c1f = (float*)carve((size_t)BATCH * HID * 4);
  float* bsum0 = (float*)carve(4096 * 4);
  float* bsum1 = (float*)carve(4096 * 4);
  float* obpad = (float*)carve(NPAD * 4);

  // -------- one-time staging --------
  cvt_bf16<<<4096, 256, 0, stream>>>(sm_W,  smW_bf,  VSZ,  INTER, KPAD, 0);
  zero_bf<<<512, 256, 0, stream>>>(smW_bf, VSZ, KPAD, INTER, KPAD - INTER);
  cvt_bf16<<<2048, 256, 0, stream>>>(out_W, outW_bf, INTER, 2560, 2560, 0);
  zero_bf<<<64, 256, 0, stream>>>(outW_bf + (size_t)INTER * 2560,
                                  NPAD - INTER, 2560, 0, 2560);
  copy_pad_f32<<<(NPAD + 255) / 256, 256, 0, stream>>>(out_b, obpad, INTER, NPAD);
  cvt_bf16<<<2048, 256, 0, stream>>>(W_ih0, Wcat0_bf, 4096, 1536, 2560, 0);
  cvt_bf16<<<2048, 256, 0, stream>>>(W_hh0, Wcat0_bf, 4096, 1024, 2560, 1536);
  cvt_bf16<<<2048, 256, 0, stream>>>(W_ih1, Wcat1_bf, 4096, 1024, 2048, 0);
  cvt_bf16<<<2048, 256, 0, stream>>>(W_hh1, Wcat1_bf, 4096, 1024, 2048, 1024);
  cvt_bf16<<<2048, 256, 0, stream>>>(enc_output, encO_bf, (long long)BATCH * SENC, ENC, ENC, 0);
  transpose_cvt<<<1024, 256, 0, stream>>>(We, WeT_bf, ENC, HID);
  transpose_cvt<<<1024, 256, 0, stream>>>(Wd, WdT_bf, HID, HID);
  vec_add<<<16, 256, 0, stream>>>(b_ih0, b_hh0, bsum0, 4096);
  vec_add<<<16, 256, 0, stream>>>(b_ih1, b_hh1, bsum1, 4096);
  embed_gather<<<2048, 256, 0, stream>>>(emb_table, y_input, embs_bf);
  init_state<<<128, 256, 0, stream>>>(dec_h0, dec_c0, h0f, c0f, h1f, c1f, h1_bf);

  // enc_proj = enc_output @ We   (2048 x 1024 x 1024)
  wmma_gemm<<<dim3(4, 64), 128, 0, stream>>>(
      encO_bf, ENC, WeT_bf, ENC, nullptr,
      enc_proj, nullptr, HID, BATCH * SENC, HID, ENC, 0);

  // -------- 63 sequential decode steps --------
  for (int t = 0; t < TSTEP; ++t) {
    const unsigned short* emb_t = embs_bf + (size_t)t * BATCH * EMB;
    float* attw_t = outF + OUT0 + (long long)t * BATCH * SENC;

    // assemble A0/A1/lin_in parts from previous state
    prep_step<<<128, 256, 0, stream>>>(emb_t, h0f, h1f, A0_bf, lin_bf, A1_bf);

    // q = h1_prev @ Wd    (32 x 1024 x 1024)
    wmma_gemm<<<dim3(4, 1), 128, 0, stream>>>(
        h1_bf, HID, WdT_bf, HID, nullptr, qbuf, nullptr, HID,
        BATCH, HID, HID, 0);

    attn_score<<<dim3(SENC, BATCH), 128, 0, stream>>>(enc_proj, qbuf, vvec, score);
    attn_softmax_ctx<<<BATCH, 256, 0, stream>>>(score, enc_mask, enc_output,
                                                attw_t, A0_bf, lin_bf);

    // LSTM0: gates = [emb|ctx|h0] @ [W_ih0|W_hh0]^T   (32 x 4096 x 2560)
    wmma_gemm<<<dim3(16, 1), 128, 0, stream>>>(
        A0_bf, 2560, Wcat0_bf, 2560, nullptr, gates, nullptr, 4096,
        BATCH, 4096, 2560, 0);
    lstm_update<<<128, 256, 0, stream>>>(gates, bsum0, c0f, h0f,
                                         A1_bf, 2048, nullptr, 0);

    // LSTM1: gates = [h0_new|h1_prev] @ [W_ih1|W_hh1]^T (32 x 4096 x 2048)
    wmma_gemm<<<dim3(16, 1), 128, 0, stream>>>(
        A1_bf, 2048, Wcat1_bf, 2048, nullptr, gates, nullptr, 4096,
        BATCH, 4096, 2048, 0);
    lstm_update<<<128, 256, 0, stream>>>(gates, bsum1, c1f, h1f,
                                         lin_bf, 2560, h1_bf, HID);

    // inter = tanh([h1|ctx|emb] @ out_W^T + out_b)  (32 x 2176 x 2560) -> bf16
    // pad rows of out_W are zero => writes tanh(0)=0 into inter_bf pad cols,
    // which is exactly the K-padding the logits GEMM needs.
    wmma_gemm<<<dim3(9, 1), 128, 0, stream>>>(
        lin_bf, 2560, outW_bf, 2560, obpad, nullptr, inter_bf, KPAD,
        BATCH, NPAD, 2560, 1);

    // logits = inter @ sm_W^T + sm_b  (32 x 32000 x 2176) -> out[b][t][:]
    wmma_gemm<<<dim3(125, 1), 128, 0, stream>>>(
        inter_bf, KPAD, smW_bf, KPAD, sm_b,
        outF + (long long)t * VSZ, nullptr, (long long)TSTEP * VSZ,
        BATCH, VSZ, KPAD, 0);
  }
}
